// LinearQuant_48687749267959
// MI455X (gfx1250) — matrix operand order repaired and checked by reading, not compiled, not run on hardware
//
#include <hip/hip_runtime.h>

// ---------------------------------------------------------------------------
// LinearQuant (x @ (int8_w * scaler)) for MI455X / gfx1250
//   M=8192 tokens, K=4096 in_features, N=4096 out_features
//   bf16 WMMA: weights (-128..127) are exact in bf16; scaler folded after GEMM.
//   Single-buffer LDS (36 KB), BK=64 -> 16 v_wmma per barrier pair.
//   All global/LDS accesses use one base address + constant immediate offsets.
// ---------------------------------------------------------------------------

typedef __attribute__((ext_vector_type(16))) __bf16 bf16x16;
typedef __attribute__((ext_vector_type(8)))  __bf16 bf16x8;
typedef __attribute__((ext_vector_type(4)))  __bf16 bf16x4;
typedef __attribute__((ext_vector_type(8)))  float  f32x8;

#define M_DIM 8192
#define K_DIM 4096
#define N_DIM 4096

#define BM 128
#define BN 128
#define BK 64
#define LDA 72   // BK + 8 pad: 72 bf16 = 144 B/row (multiple of 16 -> b128-aligned)
#define LDB 72
#define NTHREADS 256   // 8 wave32s
#define NT (K_DIM / BK) // 64 k-tiles

__device__ __forceinline__ bf16x4 cvt4(float a, float b, float c, float d) {
    bf16x4 v;
    v.x = (__bf16)a; v.y = (__bf16)b; v.z = (__bf16)c; v.w = (__bf16)d;
    return v;
}

__global__ __launch_bounds__(NTHREADS)
void linearquant_wmma_bf16(const float* __restrict__ X,
                           const int*   __restrict__ W,
                           const float* __restrict__ scaler,
                           float*       __restrict__ Out)
{
    __shared__ __bf16 As[BM * LDA];  // [m][k], k-contiguous
    __shared__ __bf16 Bs[BN * LDB];  // [n][k], k-contiguous (transposed weights)

    const int tid  = threadIdx.x;
    const int lane = tid & 31;
    const int wave = tid >> 5;
    const int wm   = wave & 3;   // 4 waves along M (4*32 = 128)
    const int wn   = wave >> 2;  // 2 waves along N (2*64 = 128)
    const int m0   = blockIdx.y * BM;
    const int n0   = blockIdx.x * BN;

    f32x8 acc[2][4];
#pragma unroll
    for (int mi = 0; mi < 2; ++mi)
#pragma unroll
        for (int ni = 0; ni < 4; ++ni)
            acc[mi][ni] = (f32x8)0.0f;

    // ---- global base pointers (computed once, advanced by constants) -------
    // A: thread owns float4 (row = tid>>4 (+16 per j), kq = tid&15); j offset
    //    = j*16 rows = j*262144 B -> 24-bit immediate.
    const float4* pA =
        (const float4*)(X + (size_t)(m0 + (tid >> 4)) * K_DIM) + (tid & 15);
    // B: thread owns col n = tid&127, k base = (tid>>7)*4; element (g,kk) at
    //    (g*8+kk)*N_DIM ints = (g*8+kk)*16384 B (max 1.0 MB) -> immediates.
    const int* pB = W + (size_t)((tid >> 7) * 4) * N_DIM + n0 + (tid & 127);

    // ---- LDS store base pointers -------------------------------------------
    __bf16* sA = &As[(tid >> 4) * LDA + (tid & 15) * 4];       // + j*16*LDA
    __bf16* sB = &Bs[(tid & 127) * LDB + (tid >> 7) * 4];      // + g*8

    // ---- LDS fragment read base pointers (ISA bf16 WMMA layouts) -----------
    // A 16x32: lanes 0-15 row=lane, K runs [0..7],[16..23]; lanes 16-31 +8.
    const __bf16* fA = &As[(wm * 32 + (lane & 15)) * LDA + ((lane >> 4) << 3)];
    // B 32x16: col = lane&15; lanes 0-15 K=[0..7],[16..23]; lanes 16-31 +8.
    const __bf16* fB = &Bs[(wn * 64 + (lane & 15)) * LDB + ((lane >> 4) << 3)];

    // ---- register staging ---------------------------------------------------
    float4 aReg[8];
    int    bReg[8][4];

    auto globalLoad = [&]() {
#pragma unroll
        for (int j = 0; j < 8; ++j)
            aReg[j] = pA[j * 16 * (K_DIM / 4)];        // imm offset j*262144 B
#pragma unroll
        for (int g = 0; g < 8; ++g)
#pragma unroll
            for (int kk = 0; kk < 4; ++kk)
                bReg[g][kk] = pB[(g * 8 + kk) * N_DIM]; // imm offset *16384 B
    };

    auto ldsStore = [&]() {
#pragma unroll
        for (int j = 0; j < 8; ++j)
            *(bf16x4*)(sA + j * 16 * LDA) =
                cvt4(aReg[j].x, aReg[j].y, aReg[j].z, aReg[j].w);
#pragma unroll
        for (int g = 0; g < 8; ++g)
            *(bf16x4*)(sB + g * 8) =
                cvt4((float)bReg[g][0], (float)bReg[g][1],
                     (float)bReg[g][2], (float)bReg[g][3]);
    };

    auto compute = [&]() {
#pragma unroll
        for (int s = 0; s < 2; ++s) {                  // two k-steps of 32
            bf16x16 afrag[2];
#pragma unroll
            for (int mi = 0; mi < 2; ++mi) {
                const __bf16* p = fA + s * 32 + mi * 16 * LDA;
                union { bf16x16 v; bf16x8 h[2]; } u;
                u.h[0] = *(const bf16x8*)(p);          // K run kb..kb+7
                u.h[1] = *(const bf16x8*)(p + 16);     // K run kb+16..kb+23
                afrag[mi] = u.v;
            }
            bf16x16 bfrag[4];
#pragma unroll
            for (int ni = 0; ni < 4; ++ni) {
                const __bf16* p = fB + s * 32 + ni * 16 * LDB;
                union { bf16x16 v; bf16x8 h[2]; } u;
                u.h[0] = *(const bf16x8*)(p);
                u.h[1] = *(const bf16x8*)(p + 16);
                bfrag[ni] = u.v;
            }
#pragma unroll
            for (int mi = 0; mi < 2; ++mi)
#pragma unroll
                for (int ni = 0; ni < 4; ++ni)
                    acc[mi][ni] = __builtin_amdgcn_wmma_f32_16x16x32_bf16(
                        false, afrag[mi], false, bfrag[ni],
                        (short)0, acc[mi][ni], false, false);
        }
    };

    // ---- pipeline: register-prefetch t+1 while computing t ------------------
    globalLoad();
    ldsStore();
    __syncthreads();

    for (int t = 0; t < NT; ++t) {
        pA += BK / 4;                 // +64 floats
        pB += (size_t)BK * N_DIM;     // +64 K-rows
        if (t + 1 < NT) globalLoad(); // long-latency loads overlap compute
        compute();
        __syncthreads();              // everyone done reading LDS tile t
        if (t + 1 < NT) ldsStore();   // overwrite with tile t+1
        __syncthreads();
    }

    // ---- epilogue: dequant-scale and store ---------------------------------
    // C/D layout: lanes 0-15 -> N=lane,    VGPR r -> M=r
    //             lanes 16-31-> N=lane-16, VGPR r -> M=8+r
    const float s     = scaler[0];
    const int   rbase = (lane >> 4) << 3;  // 0 or 8
    const int   cbase = lane & 15;
#pragma unroll
    for (int mi = 0; mi < 2; ++mi) {
#pragma unroll
        for (int ni = 0; ni < 4; ++ni) {
            const int row0 = m0 + wm * 32 + mi * 16 + rbase;
            const int col  = n0 + wn * 64 + ni * 16 + cbase;
            float* po = Out + (size_t)row0 * N_DIM + col;
#pragma unroll
            for (int r = 0; r < 8; ++r)
                po[r * N_DIM] = acc[mi][ni][r] * s;   // imm offset r*16384 B
        }
    }
}

extern "C" void kernel_launch(void* const* d_in, const int* in_sizes, int n_in,
                              void* d_out, int out_size, void* d_ws, size_t ws_size,
                              hipStream_t stream)
{
    const float* X      = (const float*)d_in[0];   // x            [8192, 4096] f32
    const float* scaler = (const float*)d_in[1];   // scaler       [1]          f32
    const int*   W      = (const int*)  d_in[2];   // int8 weights [4096, 4096] i32
    float*       Out    = (float*)d_out;           // [8192, 4096] f32

    dim3 grid(N_DIM / BN, M_DIM / BM);  // (32, 64)
    dim3 block(NTHREADS);
    linearquant_wmma_bf16<<<grid, block, 0, stream>>>(X, W, scaler, Out);
}
